// LocalEmbeddingBlock_38062000177171
// MI455X (gfx1250) — compile-verified, matrix operand order checked by reading, not run on hardware
//
#include <hip/hip_runtime.h>
#include <math.h>

// ---------------- problem constants (from reference) ----------------
#define BB      64
#define NPT     512
#define S_TOTAL (BB * NPT)   // 32768 sequences
#define KNB     16           // neighbors per sequence (tokens)
#define KFD     7            // local features
#define HID     128
#define DOUT    64
#define NH      4
#define MH      128
#define OUT_ELEMS (S_TOTAL * DOUT)   // 2,097,152 floats, then idx tail

typedef __attribute__((ext_vector_type(16))) _Float16 v16h;
typedef __attribute__((ext_vector_type(8)))  float    v8f;

#define WMMA_F16(a, b, c) \
  __builtin_amdgcn_wmma_f32_16x16x32_f16(false, (a), false, (b), (short)0, (c), false, false)

struct BlockW {
  const float *n1_g, *n1_b, *wqkv, *wo, *n2_g, *n2_b;
  const float *m_w1, *m_b1, *m_ln_g, *m_ln_b, *m_w2, *m_b2;
};
struct NetW {
  const float *w1, *b1, *ln_g, *ln_b, *w2, *b2;
  BlockW blk[2];
};

// ---------------- kernel 1: kNN (top-17 smallest sq-dist, drop self) ----
__global__ void __launch_bounds__(256)
knn_kernel(const float* __restrict__ points, float* __restrict__ idx_out) {
  int s = blockIdx.x * blockDim.x + threadIdx.x;
  if (s >= S_TOTAL) return;
  int b = s >> 9, n = s & (NPT - 1);
  const float* pb = points + (size_t)b * NPT * 2;
  float px = pb[n * 2 + 0], py = pb[n * 2 + 1];
  float dist[17];
  int   ind[17];
#pragma unroll
  for (int j = 0; j < 17; ++j) { dist[j] = 3.4e38f; ind[j] = 0; }
  for (int m = 0; m < NPT; ++m) {
    float dx = px - pb[m * 2 + 0];
    float dy = py - pb[m * 2 + 1];
    float cd = dx * dx + dy * dy;
    int   ci = m;
    // register-resident sorted insert (strict < keeps earlier index on ties,
    // matching jax.lax.top_k tie-breaking)
#pragma unroll
    for (int j = 0; j < 17; ++j) {
      if (cd < dist[j]) {
        float td = dist[j]; int ti = ind[j];
        dist[j] = cd; ind[j] = ci;
        cd = td; ci = ti;
      }
    }
  }
#pragma unroll
  for (int j = 1; j < 17; ++j)
    idx_out[(size_t)s * KNB + (j - 1)] = (float)(b * NPT + ind[j]);  // idx_flat
}

// ---------------- fragment loaders (CDNA5 WMMA VGPR layouts) ------------
// A 16x32 f16: lane<16 -> row=lane,   K = {kh..kh+7} u {16+kh..16+kh+7}, kh=0
//              lane>=16 -> row=lane-16, same with kh=8
__device__ __forceinline__ v16h ldA_f32(const float* base, int ld, int kreal) {
  int lane = threadIdx.x & 31;
  int row  = lane & 15;
  int kh   = (lane >> 4) << 3;
  v16h r;
#pragma unroll
  for (int j = 0; j < 8; ++j) {
    int k0 = kh + j, k1 = 16 + kh + j;
    r[j]     = (k0 < kreal) ? (_Float16)base[row * ld + k0] : (_Float16)0.0f;
    r[j + 8] = (k1 < kreal) ? (_Float16)base[row * ld + k1] : (_Float16)0.0f;
  }
  return r;
}
__device__ __forceinline__ v16h ldA_f16(const _Float16* base, int ld) {
  int lane = threadIdx.x & 31;
  int row  = lane & 15;
  int kh   = (lane >> 4) << 3;
  v16h r;
#pragma unroll
  for (int j = 0; j < 8; ++j) {
    r[j]     = base[row * ld + kh + j];
    r[j + 8] = base[row * ld + 16 + kh + j];
  }
  return r;
}
// B 32x16 f16 from a global f32 row-major weight W[k][n0+col], zero-pad k>=kreal
__device__ __forceinline__ v16h ldB_g(const float* W, int ncols, int kreal,
                                      int k0, int n0) {
  int lane = threadIdx.x & 31;
  int col  = (lane & 15) + n0;
  int kh   = (lane >> 4) << 3;
  v16h r;
#pragma unroll
  for (int j = 0; j < 8; ++j) {
    int ka = k0 + kh + j, kb = k0 + 16 + kh + j;
    r[j]     = (ka < kreal) ? (_Float16)W[ka * ncols + col] : (_Float16)0.0f;
    r[j + 8] = (kb < kreal) ? (_Float16)W[kb * ncols + col] : (_Float16)0.0f;
  }
  return r;
}
// A = q_head (16x16, pad K to 32) from qkv f16 LDS (stride 192)
__device__ __forceinline__ v16h ldA_q(const _Float16* qkv, int h) {
  int lane = threadIdx.x & 31;
  int row  = lane & 15;
  int kh   = (lane >> 4) << 3;
  v16h r;
#pragma unroll
  for (int j = 0; j < 8; ++j) {
    r[j]     = qkv[row * 192 + h * 16 + kh + j];   // d = kh+j in [0,16)
    r[j + 8] = (_Float16)0.0f;                     // pad K 16..31
  }
  return r;
}
// B = k^T (dh x tokens): B[d][l] = k[l][d]
__device__ __forceinline__ v16h ldB_kT(const _Float16* qkv, int h) {
  int lane = threadIdx.x & 31;
  int l    = lane & 15;
  int kh   = (lane >> 4) << 3;
  v16h r;
#pragma unroll
  for (int j = 0; j < 8; ++j) {
    r[j]     = qkv[l * 192 + 64 + h * 16 + kh + j];
    r[j + 8] = (_Float16)0.0f;
  }
  return r;
}
// B = v_head (tokens x dh): B[l][d]
__device__ __forceinline__ v16h ldB_v(const _Float16* qkv, int h) {
  int lane = threadIdx.x & 31;
  int col  = lane & 15;
  int kh   = (lane >> 4) << 3;
  v16h r;
#pragma unroll
  for (int j = 0; j < 8; ++j) {
    r[j]     = qkv[(kh + j) * 192 + 128 + h * 16 + col];
    r[j + 8] = (_Float16)0.0f;
  }
  return r;
}

// ---------------- per-wave helpers ----------------
__device__ __forceinline__ float gelu_exact(float x) {
  return 0.5f * x * (1.0f + erff(x * 0.70710678118654752f));
}
// LayerNorm over D dims for 16 rows; 2 lanes per row, combine via shfl_xor(1)
__device__ __forceinline__ void ln_f32(float* buf, int ld, int D,
                                       const float* g, const float* b) {
  int lane = threadIdx.x & 31;
  int t    = lane >> 1;
  int n    = D >> 1;
  int off  = (lane & 1) * n;
  float s = 0.f;
  for (int i = 0; i < n; ++i) s += buf[t * ld + off + i];
  s += __shfl_xor(s, 1);
  float mu = s / (float)D;
  float vs = 0.f;
  for (int i = 0; i < n; ++i) { float d = buf[t * ld + off + i] - mu; vs += d * d; }
  vs += __shfl_xor(vs, 1);
  float rstd = rsqrtf(vs / (float)D + 1e-5f);
  for (int i = 0; i < n; ++i) {
    int c = off + i;
    buf[t * ld + c] = (buf[t * ld + c] - mu) * rstd * g[c] + b[c];
  }
}
__device__ __forceinline__ void ln_f16(_Float16* buf, int ld, int D,
                                       const float* g, const float* b) {
  int lane = threadIdx.x & 31;
  int t    = lane >> 1;
  int n    = D >> 1;
  int off  = (lane & 1) * n;
  float s = 0.f;
  for (int i = 0; i < n; ++i) s += (float)buf[t * ld + off + i];
  s += __shfl_xor(s, 1);
  float mu = s / (float)D;
  float vs = 0.f;
  for (int i = 0; i < n; ++i) { float d = (float)buf[t * ld + off + i] - mu; vs += d * d; }
  vs += __shfl_xor(vs, 1);
  float rstd = rsqrtf(vs / (float)D + 1e-5f);
  for (int i = 0; i < n; ++i) {
    int c = off + i;
    float x = (float)buf[t * ld + c];
    buf[t * ld + c] = (_Float16)((x - mu) * rstd * g[c] + b[c]);
  }
}

// ---------------- kernel 2: fused embed-MLP + 2 transformer blocks ------
// One wave32 per sequence (16 tokens x 64 dims); 2 waves per block.
__global__ void __launch_bounds__(64)
fused_transformer(const float* __restrict__ feats, const float* __restrict__ mask,
                  const float* __restrict__ idx_in, float* __restrict__ out, NetW P) {
  __shared__ float    s_xs[2][16 * DOUT];     // persistent x  (4 KB/wave)
  __shared__ float    s_fs[2][16 * 192];      // f32 scratch   (12 KB/wave)
  __shared__ _Float16 s_h [2][16 * 192];      // f16 scratch (qkv / h)
  __shared__ float    s_mnb[2][16];

  const int wv   = threadIdx.x >> 5;
  const int lane = threadIdx.x & 31;
  const int seq  = blockIdx.x * 2 + wv;
  float*    xs  = s_xs[wv];
  float*    fs  = s_fs[wv];
  _Float16* hh  = s_h[wv];
  float*    mnb = s_mnb[wv];

  const int r0 = (lane >> 4) << 3;  // C-frag base row for this half-wave
  const int cc = lane & 15;         // C-frag column

  // ---- stage local pairwise features (16 tokens x 7, zero-padded to 32) ----
  if (lane < 16) {
    int gi = (int)idx_in[(size_t)seq * KNB + lane];
    float c0 = feats[(size_t)seq * 4 + 0], c1 = feats[(size_t)seq * 4 + 1];
    float c2 = feats[(size_t)seq * 4 + 2], c3 = feats[(size_t)seq * 4 + 3];
    float n0 = feats[(size_t)gi * 4 + 0], n1 = feats[(size_t)gi * 4 + 1];
    float n2 = feats[(size_t)gi * 4 + 2], n3 = feats[(size_t)gi * 4 + 3];
    float mk = mask[gi];
    float de = c0 - n0, dp = c1 - n1, d2 = c2 - n2, d3 = c3 - n3;
    float dr2 = de * de + dp * dp;
    float log_dr = 0.5f * logf(dr2 > 0.f ? dr2 : 1.f) * mk;
    float dr  = sqrtf(dr2) * mk;
    float pti = expf(c2), ptj = expf(n2);
    float kt  = fminf(pti, ptj) * dr;
    float log_kt = logf(kt > 0.f ? kt : 1.f) * mk;
    float m2  = 2.f * pti * ptj * (coshf(de) - cosf(dp));
    float log_m = logf(m2 > 0.f ? m2 : 1.f) * mk;
    float lf[KFD] = {de * mk, dp * mk, d2 * mk, d3 * mk, log_m, log_dr, log_kt};
#pragma unroll
    for (int j = 0; j < KFD; ++j) fs[lane * 32 + j] = lf[j];
#pragma unroll
    for (int j = KFD; j < 32; ++j) fs[lane * 32 + j] = 0.f;
    mnb[lane] = mk;
  }
  __syncthreads();

  // ---- embed MLP: h = gelu(lf @ w1 + b1) (K=7 padded to 32) ----
  {
    v16h a = ldA_f32(fs, 32, 32);
#pragma unroll
    for (int nt = 0; nt < 8; ++nt) {
      v16h bm = ldB_g(P.w1, HID, KFD, 0, nt * 16);
      v8f  c  = {};
      c = WMMA_F16(a, bm, c);
#pragma unroll
      for (int v = 0; v < 8; ++v) {
        int col = nt * 16 + cc;
        hh[(r0 + v) * 192 + col] = (_Float16)gelu_exact(c[v] + P.b1[col]);
      }
    }
  }
  __syncthreads();
  ln_f16(hh, 192, HID, P.ln_g, P.ln_b);
  __syncthreads();

  // ---- x = (h @ w2 + b2) * mnb ----
#pragma unroll
  for (int nt = 0; nt < 4; ++nt) {
    v8f c = {};
#pragma unroll
    for (int kt = 0; kt < 4; ++kt) {
      v16h a  = ldA_f16(hh + kt * 32, 192);
      v16h bm = ldB_g(P.w2, DOUT, HID, kt * 32, nt * 16);
      c = WMMA_F16(a, bm, c);
    }
#pragma unroll
    for (int v = 0; v < 8; ++v) {
      int col = nt * 16 + cc;
      xs[(r0 + v) * DOUT + col] = (c[v] + P.b2[col]) * mnb[r0 + v];
    }
  }
  __syncthreads();

  // ---- two transformer blocks ----
  for (int bi = 0; bi < 2; ++bi) {
    const BlockW W = P.blk[bi];

    // xn = LN(x * mnb)
    for (int i = lane; i < 16 * DOUT; i += 32) fs[i] = xs[i] * mnb[i >> 6];
    __syncthreads();
    ln_f32(fs, DOUT, DOUT, W.n1_g, W.n1_b);
    __syncthreads();

    // qkv = xn @ wqkv -> f16 LDS (16 x 192)
#pragma unroll
    for (int nt = 0; nt < 12; ++nt) {
      v8f c = {};
#pragma unroll
      for (int kt = 0; kt < 2; ++kt) {
        v16h a  = ldA_f32(fs + kt * 32, DOUT, 32);
        v16h bm = ldB_g(W.wqkv, 192, DOUT, kt * 32, nt * 16);
        c = WMMA_F16(a, bm, c);
      }
#pragma unroll
      for (int v = 0; v < 8; ++v)
        hh[(r0 + v) * 192 + nt * 16 + cc] = (_Float16)c[v];
    }
    __syncthreads();

    // attention heads: scores -> softmax (in C-frag regs) -> a @ v
    float mcol = mnb[cc];
#pragma unroll
    for (int h = 0; h < NH; ++h) {
      v16h aq = ldA_q(hh, h);
      v16h bk = ldB_kT(hh, h);
      v8f  c  = {};
      c = WMMA_F16(aq, bk, c);
#pragma unroll
      for (int v = 0; v < 8; ++v) {
        float mrow = mnb[r0 + v];
        float sc = c[v] * 0.25f + ((mrow * mcol > 0.f) ? 0.f : -1e9f);
        float mx = sc;
#pragma unroll
        for (int st = 1; st < 16; st <<= 1) mx = fmaxf(mx, __shfl_xor(mx, st));
        float p  = expf(sc - mx);
        float sm = p;
#pragma unroll
        for (int st = 1; st < 16; st <<= 1) sm += __shfl_xor(sm, st);
        c[v] = p / sm;
      }
#pragma unroll
      for (int v = 0; v < 8; ++v) fs[1024 + (r0 + v) * 16 + cc] = c[v];
      __syncthreads();
      v16h aa = ldA_f32(fs + 1024, 16, 16);
      v16h bv = ldB_v(hh, h);
      v8f  o  = {};
      o = WMMA_F16(aa, bv, o);
#pragma unroll
      for (int v = 0; v < 8; ++v)
        fs[2048 + (r0 + v) * DOUT + h * 16 + cc] = o[v];
      __syncthreads();
    }

    // x += (o @ wo) * mnb
#pragma unroll
    for (int nt = 0; nt < 4; ++nt) {
      v8f c = {};
#pragma unroll
      for (int kt = 0; kt < 2; ++kt) {
        v16h a  = ldA_f32(fs + 2048 + kt * 32, DOUT, 32);
        v16h bm = ldB_g(W.wo, DOUT, DOUT, kt * 32, nt * 16);
        c = WMMA_F16(a, bm, c);
      }
#pragma unroll
      for (int v = 0; v < 8; ++v)
        xs[(r0 + v) * DOUT + nt * 16 + cc] += c[v] * mnb[r0 + v];
    }
    __syncthreads();

    // xn2 = LN(x)
    for (int i = lane; i < 16 * DOUT; i += 32) fs[i] = xs[i];
    __syncthreads();
    ln_f32(fs, DOUT, DOUT, W.n2_g, W.n2_b);
    __syncthreads();

    // h = gelu(xn2 @ m_w1 + m_b1)
#pragma unroll
    for (int nt = 0; nt < 8; ++nt) {
      v8f c = {};
#pragma unroll
      for (int kt = 0; kt < 2; ++kt) {
        v16h a  = ldA_f32(fs + kt * 32, DOUT, 32);
        v16h bm = ldB_g(W.m_w1, MH, DOUT, kt * 32, nt * 16);
        c = WMMA_F16(a, bm, c);
      }
#pragma unroll
      for (int v = 0; v < 8; ++v) {
        int col = nt * 16 + cc;
        hh[(r0 + v) * 192 + col] = (_Float16)gelu_exact(c[v] + W.m_b1[col]);
      }
    }
    __syncthreads();
    ln_f16(hh, 192, MH, W.m_ln_g, W.m_ln_b);
    __syncthreads();

    // x += (h @ m_w2 + m_b2) * mnb
#pragma unroll
    for (int nt = 0; nt < 4; ++nt) {
      v8f c = {};
#pragma unroll
      for (int kt = 0; kt < 4; ++kt) {
        v16h a  = ldA_f16(hh + kt * 32, 192);
        v16h bm = ldB_g(W.m_w2, DOUT, MH, kt * 32, nt * 16);
        c = WMMA_F16(a, bm, c);
      }
#pragma unroll
      for (int v = 0; v < 8; ++v) {
        int col = nt * 16 + cc;
        xs[(r0 + v) * DOUT + col] += (c[v] + W.m_b2[col]) * mnb[r0 + v];
      }
    }
    __syncthreads();
  }

  // ---- masked mean over the 16 tokens ----
  float denom = 0.f;
#pragma unroll
  for (int k = 0; k < KNB; ++k) denom += 1e-9f + mnb[k];
  float mc = mask[seq];
#pragma unroll
  for (int t = 0; t < 2; ++t) {
    int col = lane + t * 32;
    float ssum = 0.f;
#pragma unroll
    for (int r = 0; r < 16; ++r) ssum += xs[r * DOUT + col];
    out[(size_t)seq * DOUT + col] = ssum / denom * mc;
  }
}

// ---------------- host launch ----------------
extern "C" void kernel_launch(void* const* d_in, const int* in_sizes, int n_in,
                              void* d_out, int out_size, void* d_ws, size_t ws_size,
                              hipStream_t stream) {
  (void)in_sizes; (void)n_in; (void)out_size; (void)d_ws; (void)ws_size;
  // setup_inputs() insertion order: points, features, mask, then params
  // flattened in insertion order: mlp{w1,b1,ln_g,ln_b,w2,b2},
  // blocks[0..1]{n1_g,n1_b,wqkv,wo,n2_g,n2_b,m_w1,m_b1,m_ln_g,m_ln_b,m_w2,m_b2}
  const float* points = (const float*)d_in[0];
  const float* feats  = (const float*)d_in[1];
  const float* maskp  = (const float*)d_in[2];
  NetW P;
  int i = 3;
  P.w1   = (const float*)d_in[i++];
  P.b1   = (const float*)d_in[i++];
  P.ln_g = (const float*)d_in[i++];
  P.ln_b = (const float*)d_in[i++];
  P.w2   = (const float*)d_in[i++];
  P.b2   = (const float*)d_in[i++];
  for (int b = 0; b < 2; ++b) {
    P.blk[b].n1_g   = (const float*)d_in[i++];
    P.blk[b].n1_b   = (const float*)d_in[i++];
    P.blk[b].wqkv   = (const float*)d_in[i++];
    P.blk[b].wo     = (const float*)d_in[i++];
    P.blk[b].n2_g   = (const float*)d_in[i++];
    P.blk[b].n2_b   = (const float*)d_in[i++];
    P.blk[b].m_w1   = (const float*)d_in[i++];
    P.blk[b].m_b1   = (const float*)d_in[i++];
    P.blk[b].m_ln_g = (const float*)d_in[i++];
    P.blk[b].m_ln_b = (const float*)d_in[i++];
    P.blk[b].m_w2   = (const float*)d_in[i++];
    P.blk[b].m_b2   = (const float*)d_in[i++];
  }
  float* out  = (float*)d_out;
  float* idxf = out + OUT_ELEMS;   // second tuple output: idx_flat (as float)

  knn_kernel<<<S_TOTAL / 256, 256, 0, stream>>>(points, idxf);
  fused_transformer<<<S_TOTAL / 2, 64, 0, stream>>>(feats, maskp, idxf, out, P);
}
